// MultiAgentMHCModel_53102975647876
// MI455X (gfx1250) — compile-verified, matrix operand order checked by reading, not run on hardware
//
#include <hip/hip_runtime.h>

// ---------------------------------------------------------------------------
// MultiAgentMHC on MI455X (gfx1250, wave32).
// Big GEMMs: WMMA pipe via split-bf16 (hi/lo) with f32 accumulate:
//   A*B ~= Ah*Bh + Ah*Bl + Al*Bh   (3x v_wmma_f32_16x16x32_bf16 per tile-K)
// QKV GEMMs additionally use pre-converted planar bf16 operands staged into
// LDS by the Tensor Data Mover (tensor_load_to_lds + s_wait_tensorcnt), with
// the D# pad feature creating the padded [row][40] LDS layout in hardware.
// ---------------------------------------------------------------------------

typedef __attribute__((ext_vector_type(16))) __bf16 v16bf;
typedef __attribute__((ext_vector_type(8)))  __bf16 v8bf;
typedef __attribute__((ext_vector_type(8)))  float  v8f;
typedef __attribute__((ext_vector_type(4)))  float  v4f;
typedef __attribute__((ext_vector_type(4)))  unsigned v4u;
typedef __attribute__((ext_vector_type(8)))  unsigned v8u;

union Frag16 { v16bf v; v8bf h[2]; };

#define KPAD 40   // 32 data + 8 pad bf16 per LDS row; 80B stride (16B aligned)

// split f32 -> bf16 hi (truncate) + bf16 lo (residual)
__device__ __forceinline__ void split2(float x, unsigned short* hi, unsigned short* lo) {
  unsigned u = __float_as_uint(x);
  unsigned short hb = (unsigned short)(u >> 16);
  float r = x - __uint_as_float((unsigned)hb << 16);
  *hi = hb;
  *lo = (unsigned short)(__float_as_uint(r) >> 16);
}

__device__ __forceinline__ v8bf ld8(const unsigned short* p) {
  return *(const v8bf*)p;
}

__device__ __forceinline__ float block_reduce_sum(float v, float* red, int t) {
  red[t] = v; __syncthreads();
  #pragma unroll
  for (int off = 128; off > 0; off >>= 1) {
    if (t < off) red[t] += red[t + off];
    __syncthreads();
  }
  float r = red[0];
  __syncthreads();
  return r;
}

// ---------------------------------------------------------------------------
// TDM: 2D tile load (bf16 elements) global -> LDS, with LDS row padding:
// 64B of data then 16B pad (pad_interval code 3 = 16 DWORDs, pad_amount
// code 3 = 4 DWORDs) -> matches the ushort [rows][KPAD=40] LDS layout.
// 2-group descriptor form (2D tensor, VADDR2/3 = NULL).
// ---------------------------------------------------------------------------
__device__ __forceinline__ void tdm_load_2d(unsigned lds_addr, const void* gptr,
                                            unsigned tile_d0, unsigned tile_d1,
                                            unsigned long long row_pitch_elems,
                                            unsigned tensor_d0, unsigned tensor_d1)
{
  unsigned long long ga = (unsigned long long)gptr;
  v4u g0;
  g0[0] = 1u;                                       // count=1, user descriptor
  g0[1] = lds_addr;                                 // LDS byte address
  g0[2] = (unsigned)ga;                             // global_addr[31:0]
  g0[3] = (unsigned)((ga >> 32) & 0x01FFFFFFull)    // global_addr[56:32]
        | (2u << 30);                               // type=2 ("image")
  v8u g1;
  g1[0] = (1u << 16)                                // data_size: 2 bytes
        | (1u << 20)                                // pad_enable
        | (3u << 22)                                // pad_interval: 16 DWORDs
        | (3u << 25);                               // pad_amount: 4 DWORDs
  g1[1] = (tensor_d0 & 0xFFFFu) << 16;              // tensor_dim0[15:0]
  g1[2] = (tensor_d0 >> 16) | ((tensor_d1 & 0xFFFFu) << 16);
  g1[3] = (tensor_d1 >> 16) | (tile_d0 << 16);      // tile_dim0
  g1[4] = tile_d1 & 0xFFFFu;                        // tile_dim1 (tile_dim2=0)
  g1[5] = (unsigned)row_pitch_elems;                // dim0_stride[31:0]
  g1[6] = (unsigned)((row_pitch_elems >> 32) & 0xFFFFull);
  g1[7] = 0u;
  asm volatile("tensor_load_to_lds %0, %1" :: "s"(g0), "s"(g1) : "memory");
}

// ---------------------------------------------------------------------------
// Generic WMMA GEMM (f32 in, VALU split staging):
//   C[z] = act(A[z] (MxK) @ W[z] (KxN) + bias[z])
// macro tile 128x64, K tile 32; 8 waves, each owns 32x32 (2x2 WMMA tiles).
// grid = (N/64, M/128, Z). Used for input projection and per-agent heads.
// ---------------------------------------------------------------------------
template<int DO_GELU>
__global__ __launch_bounds__(256)
void gemm_kernel(const float* __restrict__ Abase, long lda, long az,
                 const float* __restrict__ Wbase, long ldw, long wz,
                 const float* __restrict__ biasbase, long bz,
                 float* __restrict__ Cbase, long ldc, long cz,
                 int K)
{
  __shared__ __align__(16) unsigned short sAh[128][KPAD];
  __shared__ __align__(16) unsigned short sAl[128][KPAD];
  __shared__ __align__(16) unsigned short sBh[64][KPAD];
  __shared__ __align__(16) unsigned short sBl[64][KPAD];

  const int z = blockIdx.z;
  const float* A    = Abase    + (size_t)z * az;
  const float* W    = Wbase    + (size_t)z * wz;
  const float* bias = biasbase + (size_t)z * bz;
  float*       C    = Cbase    + (size_t)z * cz;

  const int t    = threadIdx.x;
  const int lane = t & 31;
  const int wid  = t >> 5;
  const int wm   = wid & 3;
  const int wn   = wid >> 2;
  const int m0   = blockIdx.y * 128;
  const int n0   = blockIdx.x * 64;
  const int hlf  = lane >> 4;
  const int l16  = lane & 15;

  const v8f vzero = {};
  v8f acc[2][2];
  #pragma unroll
  for (int i = 0; i < 2; i++)
    #pragma unroll
    for (int j = 0; j < 2; j++)
      acc[i][j] = vzero;

  for (int k0 = 0; k0 < K; k0 += 32) {
    #pragma unroll
    for (int i = 0; i < 4; i++) {
      int vi = i * 256 + t;
      int r  = vi >> 3;
      int c4 = (vi & 7) << 2;
      v4f val = *(const v4f*)(A + (size_t)(m0 + r) * lda + (k0 + c4));
      #pragma unroll
      for (int e = 0; e < 4; e++)
        split2(val[e], &sAh[r][c4 + e], &sAl[r][c4 + e]);
    }
    #pragma unroll
    for (int i = 0; i < 2; i++) {
      int vi = i * 256 + t;
      int r  = vi >> 4;
      int c4 = (vi & 15) << 2;
      v4f val = *(const v4f*)(W + (size_t)(k0 + r) * ldw + (n0 + c4));
      #pragma unroll
      for (int e = 0; e < 4; e++)
        split2(val[e], &sBh[c4 + e][r], &sBl[c4 + e][r]);
    }
    __syncthreads();

    Frag16 ah[2], al[2], bh2[2], bl2[2];
    #pragma unroll
    for (int i = 0; i < 2; i++) {
      const unsigned short* pa  = &sAh[wm * 32 + i * 16 + l16][hlf * 8];
      const unsigned short* pal = &sAl[wm * 32 + i * 16 + l16][hlf * 8];
      ah[i].h[0] = ld8(pa);   ah[i].h[1] = ld8(pa + 16);
      al[i].h[0] = ld8(pal);  al[i].h[1] = ld8(pal + 16);
      const unsigned short* pb  = &sBh[wn * 32 + i * 16 + l16][hlf * 8];
      const unsigned short* pbl = &sBl[wn * 32 + i * 16 + l16][hlf * 8];
      bh2[i].h[0] = ld8(pb);  bh2[i].h[1] = ld8(pb + 16);
      bl2[i].h[0] = ld8(pbl); bl2[i].h[1] = ld8(pbl + 16);
    }
    #pragma unroll
    for (int i = 0; i < 2; i++)
      #pragma unroll
      for (int j = 0; j < 2; j++) {
        acc[i][j] = __builtin_amdgcn_wmma_f32_16x16x32_bf16(false, ah[i].v, false, bh2[j].v, (short)0, acc[i][j], false, false);
        acc[i][j] = __builtin_amdgcn_wmma_f32_16x16x32_bf16(false, ah[i].v, false, bl2[j].v, (short)0, acc[i][j], false, false);
        acc[i][j] = __builtin_amdgcn_wmma_f32_16x16x32_bf16(false, al[i].v, false, bh2[j].v, (short)0, acc[i][j], false, false);
      }
    __syncthreads();
  }

  #pragma unroll
  for (int i = 0; i < 2; i++) {
    #pragma unroll
    for (int j = 0; j < 2; j++) {
      int   ncol = n0 + wn * 32 + j * 16 + l16;
      float bvv  = bias[ncol];
      #pragma unroll
      for (int r = 0; r < 8; r++) {
        int   mrow = m0 + wm * 32 + i * 16 + hlf * 8 + r;
        float x    = acc[i][j][r] + bvv;
        if (DO_GELU) x = 0.5f * x * (1.0f + erff(x * 0.70710678118654752f));
        C[(size_t)mrow * ldc + ncol] = x;
      }
    }
  }
}

// ---------------------------------------------------------------------------
// QKV GEMM: planar bf16 hi/lo inputs, TDM-staged LDS tiles.
//   A  : [Mtot][K] bf16 (hi, lo planes)  -- produced by ln_kernel
//   Wt : [Ntot][K] bf16 (hi, lo planes)  -- pre-transposed by wconv_kernel
//   C  : f32 [Mtot][ldc], + bias
// grid = (Ntot/64, Mtot/128). Wave 0 drives the Tensor Data Mover.
// ---------------------------------------------------------------------------
__global__ __launch_bounds__(256)
void gemm_bf_tdm_kernel(const unsigned short* __restrict__ Ah,
                        const unsigned short* __restrict__ Al,
                        const unsigned short* __restrict__ Wth,
                        const unsigned short* __restrict__ Wtl,
                        const float* __restrict__ bias,
                        float* __restrict__ C, long ldc,
                        int K, int Mtot, int Ntot)
{
  __shared__ __align__(16) unsigned short sAh[128][KPAD];
  __shared__ __align__(16) unsigned short sAl[128][KPAD];
  __shared__ __align__(16) unsigned short sBh[64][KPAD];
  __shared__ __align__(16) unsigned short sBl[64][KPAD];

  const int t    = threadIdx.x;
  const int lane = t & 31;
  const int wid  = t >> 5;
  const int wm   = wid & 3;
  const int wn   = wid >> 2;
  const int m0   = blockIdx.y * 128;
  const int n0   = blockIdx.x * 64;
  const int hlf  = lane >> 4;
  const int l16  = lane & 15;

  const unsigned ldsAh = (unsigned)(unsigned long long)(uintptr_t)&sAh[0][0];
  const unsigned ldsAl = (unsigned)(unsigned long long)(uintptr_t)&sAl[0][0];
  const unsigned ldsBh = (unsigned)(unsigned long long)(uintptr_t)&sBh[0][0];
  const unsigned ldsBl = (unsigned)(unsigned long long)(uintptr_t)&sBl[0][0];

  const v8f vzero = {};
  v8f acc[2][2];
  #pragma unroll
  for (int i = 0; i < 2; i++)
    #pragma unroll
    for (int j = 0; j < 2; j++)
      acc[i][j] = vzero;

  for (int k0 = 0; k0 < K; k0 += 32) {
    if (wid == 0) {
      // 4 DMA tiles: A hi/lo 128x32, B hi/lo 64x32 (all rows K-pitched)
      tdm_load_2d(ldsAh, Ah + (size_t)m0 * K + k0, 32u, 128u,
                  (unsigned long long)K, (unsigned)K, (unsigned)Mtot);
      tdm_load_2d(ldsAl, Al + (size_t)m0 * K + k0, 32u, 128u,
                  (unsigned long long)K, (unsigned)K, (unsigned)Mtot);
      tdm_load_2d(ldsBh, Wth + (size_t)n0 * K + k0, 32u, 64u,
                  (unsigned long long)K, (unsigned)K, (unsigned)Ntot);
      tdm_load_2d(ldsBl, Wtl + (size_t)n0 * K + k0, 32u, 64u,
                  (unsigned long long)K, (unsigned)K, (unsigned)Ntot);
      __builtin_amdgcn_s_wait_tensorcnt(0);
    }
    __syncthreads();

    Frag16 ah[2], al[2], bh2[2], bl2[2];
    #pragma unroll
    for (int i = 0; i < 2; i++) {
      const unsigned short* pa  = &sAh[wm * 32 + i * 16 + l16][hlf * 8];
      const unsigned short* pal = &sAl[wm * 32 + i * 16 + l16][hlf * 8];
      ah[i].h[0] = ld8(pa);   ah[i].h[1] = ld8(pa + 16);
      al[i].h[0] = ld8(pal);  al[i].h[1] = ld8(pal + 16);
      const unsigned short* pb  = &sBh[wn * 32 + i * 16 + l16][hlf * 8];
      const unsigned short* pbl = &sBl[wn * 32 + i * 16 + l16][hlf * 8];
      bh2[i].h[0] = ld8(pb);  bh2[i].h[1] = ld8(pb + 16);
      bl2[i].h[0] = ld8(pbl); bl2[i].h[1] = ld8(pbl + 16);
    }
    #pragma unroll
    for (int i = 0; i < 2; i++)
      #pragma unroll
      for (int j = 0; j < 2; j++) {
        acc[i][j] = __builtin_amdgcn_wmma_f32_16x16x32_bf16(false, ah[i].v, false, bh2[j].v, (short)0, acc[i][j], false, false);
        acc[i][j] = __builtin_amdgcn_wmma_f32_16x16x32_bf16(false, ah[i].v, false, bl2[j].v, (short)0, acc[i][j], false, false);
        acc[i][j] = __builtin_amdgcn_wmma_f32_16x16x32_bf16(false, al[i].v, false, bh2[j].v, (short)0, acc[i][j], false, false);
      }
    __syncthreads();
  }

  #pragma unroll
  for (int i = 0; i < 2; i++) {
    #pragma unroll
    for (int j = 0; j < 2; j++) {
      int   ncol = n0 + wn * 32 + j * 16 + l16;
      float bvv  = bias[ncol];
      #pragma unroll
      for (int r = 0; r < 8; r++) {
        int mrow = m0 + wm * 32 + i * 16 + hlf * 8 + r;
        C[(size_t)mrow * ldc + ncol] = acc[i][j][r] + bvv;
      }
    }
  }
}

// ---------------------------------------------------------------------------
// Weight transpose + f32->bf16 hi/lo split: W[z] (KxN) -> Wt[z] (NxK) planes.
// grid = (N/32, K/32, Z), 256 threads, LDS 32x33 tile.
// ---------------------------------------------------------------------------
__global__ __launch_bounds__(256)
void wconv_kernel(const float* __restrict__ W, unsigned short* __restrict__ Wh,
                  unsigned short* __restrict__ Wl, int K, int N)
{
  __shared__ float tile[32][33];
  const int z = blockIdx.z;
  const float* Wz = W + (size_t)z * K * N;
  unsigned short* Whz = Wh + (size_t)z * K * N;
  unsigned short* Wlz = Wl + (size_t)z * K * N;
  const int n0 = blockIdx.x * 32, k0 = blockIdx.y * 32;
  const int t = threadIdx.x;
  const int c = t & 31, rr = t >> 5;
  #pragma unroll
  for (int i = 0; i < 4; i++) {
    int r = i * 8 + rr;
    tile[r][c] = Wz[(size_t)(k0 + r) * N + (n0 + c)];
  }
  __syncthreads();
  #pragma unroll
  for (int i = 0; i < 4; i++) {
    int r = i * 8 + rr;          // out row n0+r, out col k0+c
    size_t o = (size_t)(n0 + r) * K + (k0 + c);
    split2(tile[c][r], &Whz[o], &Wlz[o]);
  }
}

// ---------------------------------------------------------------------------
// scores[b] = q[b] (64x1024) @ k[b]^T (1024x64) * scale  (f32 in, WMMA).
// One block (4 waves) per batch; wave grid 2x2 of 32x32 blocks.
// ---------------------------------------------------------------------------
__global__ __launch_bounds__(128)
void scores_kernel(const float* __restrict__ qg, const float* __restrict__ kg,
                   float* __restrict__ attn, float scale)
{
  __shared__ __align__(16) unsigned short sQh[64][KPAD];
  __shared__ __align__(16) unsigned short sQl[64][KPAD];
  __shared__ __align__(16) unsigned short sKh[64][KPAD];
  __shared__ __align__(16) unsigned short sKl[64][KPAD];

  const int b = blockIdx.x;
  const float* Q  = qg + (size_t)b * 64 * 1024;
  const float* Kp = kg + (size_t)b * 64 * 1024;

  const int t    = threadIdx.x;
  const int lane = t & 31;
  const int wid  = t >> 5;
  const int wm   = wid & 1;
  const int wn   = wid >> 1;
  const int hlf  = lane >> 4;
  const int l16  = lane & 15;

  const v8f vzero = {};
  v8f acc[2][2];
  #pragma unroll
  for (int i = 0; i < 2; i++)
    #pragma unroll
    for (int j = 0; j < 2; j++)
      acc[i][j] = vzero;

  for (int k0 = 0; k0 < 1024; k0 += 32) {
    #pragma unroll
    for (int i = 0; i < 4; i++) {
      int vi = i * 128 + t;
      int r  = vi >> 3;
      int c4 = (vi & 7) << 2;
      v4f vq = *(const v4f*)(Q  + (size_t)r * 1024 + k0 + c4);
      v4f vk = *(const v4f*)(Kp + (size_t)r * 1024 + k0 + c4);
      #pragma unroll
      for (int e = 0; e < 4; e++) {
        split2(vq[e], &sQh[r][c4 + e], &sQl[r][c4 + e]);
        split2(vk[e], &sKh[r][c4 + e], &sKl[r][c4 + e]);
      }
    }
    __syncthreads();

    Frag16 ah[2], al[2], bh2[2], bl2[2];
    #pragma unroll
    for (int i = 0; i < 2; i++) {
      const unsigned short* pa  = &sQh[wm * 32 + i * 16 + l16][hlf * 8];
      const unsigned short* pal = &sQl[wm * 32 + i * 16 + l16][hlf * 8];
      ah[i].h[0] = ld8(pa);   ah[i].h[1] = ld8(pa + 16);
      al[i].h[0] = ld8(pal);  al[i].h[1] = ld8(pal + 16);
      const unsigned short* pb  = &sKh[wn * 32 + i * 16 + l16][hlf * 8];
      const unsigned short* pbl = &sKl[wn * 32 + i * 16 + l16][hlf * 8];
      bh2[i].h[0] = ld8(pb);  bh2[i].h[1] = ld8(pb + 16);
      bl2[i].h[0] = ld8(pbl); bl2[i].h[1] = ld8(pbl + 16);
    }
    #pragma unroll
    for (int i = 0; i < 2; i++)
      #pragma unroll
      for (int j = 0; j < 2; j++) {
        acc[i][j] = __builtin_amdgcn_wmma_f32_16x16x32_bf16(false, ah[i].v, false, bh2[j].v, (short)0, acc[i][j], false, false);
        acc[i][j] = __builtin_amdgcn_wmma_f32_16x16x32_bf16(false, ah[i].v, false, bl2[j].v, (short)0, acc[i][j], false, false);
        acc[i][j] = __builtin_amdgcn_wmma_f32_16x16x32_bf16(false, al[i].v, false, bh2[j].v, (short)0, acc[i][j], false, false);
      }
    __syncthreads();
  }

  float* S = attn + (size_t)b * 4096;
  #pragma unroll
  for (int i = 0; i < 2; i++)
    #pragma unroll
    for (int j = 0; j < 2; j++) {
      int ncol = wn * 32 + j * 16 + l16;
      #pragma unroll
      for (int r = 0; r < 8; r++) {
        int mrow = wm * 32 + i * 16 + hlf * 8 + r;
        S[(size_t)mrow * 64 + ncol] = acc[i][j][r] * scale;
      }
    }
}

// ---------------------------------------------------------------------------
// Sinkhorn: 50 alternating row/col log-space normalizations on 64x64 in
// padded LDS, then exp. One block (256 thr) per batch; shfl_xor reductions.
// ---------------------------------------------------------------------------
__global__ __launch_bounds__(256)
void sinkhorn_kernel(float* __restrict__ attn)
{
  __shared__ float la[64][65];
  const int b = blockIdx.x, t = threadIdx.x;
  float* S = attn + (size_t)b * 4096;
  #pragma unroll
  for (int i = 0; i < 16; i++) {
    int idx = i * 256 + t;
    la[idx >> 6][idx & 63] = S[idx];
  }
  __syncthreads();
  const int g = t >> 2, p = t & 3;
  for (int it = 0; it < 50; it++) {
    float m = -3.0e38f;
    #pragma unroll
    for (int j = 0; j < 16; j++) m = fmaxf(m, la[g][p * 16 + j]);
    m = fmaxf(m, __shfl_xor(m, 1, 32));
    m = fmaxf(m, __shfl_xor(m, 2, 32));
    float s = 0.0f;
    #pragma unroll
    for (int j = 0; j < 16; j++) s += __expf(la[g][p * 16 + j] - m);
    s += __shfl_xor(s, 1, 32);
    s += __shfl_xor(s, 2, 32);
    float lse = m + __logf(s);
    #pragma unroll
    for (int j = 0; j < 16; j++) la[g][p * 16 + j] -= lse;
    __syncthreads();
    m = -3.0e38f;
    #pragma unroll
    for (int j = 0; j < 16; j++) m = fmaxf(m, la[p * 16 + j][g]);
    m = fmaxf(m, __shfl_xor(m, 1, 32));
    m = fmaxf(m, __shfl_xor(m, 2, 32));
    s = 0.0f;
    #pragma unroll
    for (int j = 0; j < 16; j++) s += __expf(la[p * 16 + j][g] - m);
    s += __shfl_xor(s, 1, 32);
    s += __shfl_xor(s, 2, 32);
    lse = m + __logf(s);
    #pragma unroll
    for (int j = 0; j < 16; j++) la[p * 16 + j][g] -= lse;
    __syncthreads();
  }
  #pragma unroll
  for (int i = 0; i < 16; i++) {
    int idx = i * 256 + t;
    S[idx] = __expf(la[idx >> 6][idx & 63]);
  }
}

// ---------------------------------------------------------------------------
// Fused: attended = attn[b] @ v[b];  mixed = 0.1*ns + 0.9*attended;
// scaled = (mixed + ab[a]) * sf[a];  h += scaled / max(1, ||scaled||_row).
// One block per (b,a) row; each thread owns 4 of H=1024 columns.
// ---------------------------------------------------------------------------
__global__ __launch_bounds__(256)
void attend_mix_kernel(float* __restrict__ h, const float* __restrict__ ns,
                       const float* __restrict__ attn, const float* __restrict__ vg,
                       const float* __restrict__ ab, const float* __restrict__ sf)
{
  __shared__ float arow[64];
  __shared__ float red[256];
  const int rid = blockIdx.x;
  const int b = rid >> 6, a = rid & 63;
  const int t = threadIdx.x;
  if (t < 64) arow[t] = attn[(size_t)b * 4096 + (size_t)a * 64 + t];
  __syncthreads();

  float acc[4] = {0.f, 0.f, 0.f, 0.f};
  const float* vb = vg + (size_t)b * 64 * 1024;
  for (int c = 0; c < 64; c++) {
    float w = arow[c];
    const float* vr = vb + (size_t)c * 1024;
    #pragma unroll
    for (int j = 0; j < 4; j++) acc[j] += w * vr[t + j * 256];
  }

  const float* nsr = ns + (size_t)rid * 1024;
  const float* abr = ab + (size_t)a * 1024;
  const float sfa = sf[a];
  float sc[4];
  float ss = 0.f;
  #pragma unroll
  for (int j = 0; j < 4; j++) {
    int col = t + j * 256;
    float mixed = 0.1f * nsr[col] + 0.9f * acc[j];
    float s = (mixed + abr[col]) * sfa;
    sc[j] = s;
    ss += s * s;
  }
  float norm = sqrtf(block_reduce_sum(ss, red, t));
  float inv = 1.0f / fmaxf(1.0f, norm);   // SIG_BOUND = 1
  float* hr = h + (size_t)rid * 1024;
  #pragma unroll
  for (int j = 0; j < 4; j++) hr[t + j * 256] += sc[j] * inv;
}

// ---------------------------------------------------------------------------
// LayerNorm over H=1024 per row (in-place safe). Optionally also emits the
// normalized row as planar bf16 hi/lo (feeds the TDM-staged QKV GEMMs).
// ---------------------------------------------------------------------------
__global__ __launch_bounds__(256)
void ln_kernel(const float* x, float* y,
               const float* __restrict__ gw, const float* __restrict__ bw,
               unsigned short* yh, unsigned short* yl)
{
  __shared__ float red[256];
  const int rid = blockIdx.x, t = threadIdx.x;
  const float* xr = x + (size_t)rid * 1024;
  float vbuf[4];
  float s = 0.f;
  #pragma unroll
  for (int j = 0; j < 4; j++) { vbuf[j] = xr[t + j * 256]; s += vbuf[j]; }
  float mu = block_reduce_sum(s, red, t) * (1.0f / 1024.0f);
  float ss = 0.f;
  #pragma unroll
  for (int j = 0; j < 4; j++) { float d = vbuf[j] - mu; ss += d * d; }
  float var = block_reduce_sum(ss, red, t) * (1.0f / 1024.0f);
  float inv = rsqrtf(var + 1e-5f);
  float* yr = y + (size_t)rid * 1024;
  #pragma unroll
  for (int j = 0; j < 4; j++) {
    int col = t + j * 256;
    float o = (vbuf[j] - mu) * inv * gw[col] + bw[col];
    yr[col] = o;
    if (yh) {
      size_t off = (size_t)rid * 1024 + col;
      split2(o, &yh[off], &yl[off]);
    }
  }
}

// ---------------------------------------------------------------------------
extern "C" void kernel_launch(void* const* d_in, const int* in_sizes, int n_in,
                              void* d_out, int out_size, void* d_ws, size_t ws_size,
                              hipStream_t stream)
{
  (void)in_sizes; (void)n_in; (void)out_size; (void)ws_size;

  const float* X     = (const float*)d_in[0];   // agent_states [256,64,512]
  const float* Win   = (const float*)d_in[1];   // [512,1024]
  const float* bin_  = (const float*)d_in[2];   // [1024]
  const float* ln_g  = (const float*)d_in[3];   // [3,1024]
  const float* ln_b  = (const float*)d_in[4];   // [3,1024]
  const float* Wq    = (const float*)d_in[5];   // [3,1024,1024]
  const float* bq    = (const float*)d_in[6];   // [3,1024]
  const float* Wk    = (const float*)d_in[7];
  const float* bk    = (const float*)d_in[8];
  const float* Wv    = (const float*)d_in[9];
  const float* bv    = (const float*)d_in[10];
  const float* ab    = (const float*)d_in[11];  // [3,64,1024]
  const float* sf    = (const float*)d_in[12];  // [3,64]
  const float* iln_g = (const float*)d_in[13];  // [2,1024]
  const float* iln_b = (const float*)d_in[14];  // [2,1024]
  const float* W1    = (const float*)d_in[15];  // [64,1024,512]
  const float* b1    = (const float*)d_in[16];  // [64,512]
  const float* W2    = (const float*)d_in[17];  // [64,512,256]
  const float* b2    = (const float*)d_in[18];  // [64,256]
  float* out = (float*)d_out;                   // [256,64,256]

  const size_t MB = 1024ull * 1024ull;
  char* ws = (char*)d_ws;
  float*          h    = (float*)(ws + 0 * MB);      // 64 MB
  float*          ns   = (float*)(ws + 64 * MB);     // 64 MB
  float*          q    = (float*)(ws + 128 * MB);    // 64 MB (reused as y)
  float*          kk   = (float*)(ws + 192 * MB);    // 64 MB
  float*          vv   = (float*)(ws + 256 * MB);    // 64 MB
  float*          attn = (float*)(ws + 320 * MB);    // 4 MB
  unsigned short* nsh  = (unsigned short*)(ws + 324 * MB);  // 32 MB
  unsigned short* nsl  = (unsigned short*)(ws + 356 * MB);  // 32 MB
  unsigned short* Wqth = (unsigned short*)(ws + 388 * MB);  // 6 MB
  unsigned short* Wqtl = (unsigned short*)(ws + 394 * MB);
  unsigned short* Wkth = (unsigned short*)(ws + 400 * MB);
  unsigned short* Wktl = (unsigned short*)(ws + 406 * MB);
  unsigned short* Wvth = (unsigned short*)(ws + 412 * MB);
  unsigned short* Wvtl = (unsigned short*)(ws + 418 * MB);  // end 424 MB
  float* y = q;

  const dim3 blk256(256);
  const dim3 blk128(128);
  const int M = 16384;   // B*A rows

  // ---- one-time per call: transpose+split all QKV weights to bf16 planes
  wconv_kernel<<<dim3(32, 32, 3), blk256, 0, stream>>>(Wq, Wqth, Wqtl, 1024, 1024);
  wconv_kernel<<<dim3(32, 32, 3), blk256, 0, stream>>>(Wk, Wkth, Wktl, 1024, 1024);
  wconv_kernel<<<dim3(32, 32, 3), blk256, 0, stream>>>(Wv, Wvth, Wvtl, 1024, 1024);

  // ---- input projection: h = X @ Win + bin_   (16384x512x1024, f32 path)
  gemm_kernel<0><<<dim3(16, M / 128, 1), blk256, 0, stream>>>(
      X, 512, 0, Win, 1024, 0, bin_, 0, h, 1024, 0, 512);

  for (int i = 0; i < 3; i++) {
    // ns = LN(h)  (+ bf16 hi/lo planes for the TDM GEMMs)
    ln_kernel<<<M, blk256, 0, stream>>>(h, ns, ln_g + i * 1024, ln_b + i * 1024,
                                        nsh, nsl);

    // q,k,v = ns @ W{q,k,v}[i] + b  (16384x1024x1024, TDM + WMMA path)
    gemm_bf_tdm_kernel<<<dim3(16, M / 128), blk256, 0, stream>>>(
        nsh, nsl, Wqth + (size_t)i * 1024 * 1024, Wqtl + (size_t)i * 1024 * 1024,
        bq + i * 1024, q, 1024, 1024, M, 1024);
    gemm_bf_tdm_kernel<<<dim3(16, M / 128), blk256, 0, stream>>>(
        nsh, nsl, Wkth + (size_t)i * 1024 * 1024, Wktl + (size_t)i * 1024 * 1024,
        bk + i * 1024, kk, 1024, 1024, M, 1024);
    gemm_bf_tdm_kernel<<<dim3(16, M / 128), blk256, 0, stream>>>(
        nsh, nsl, Wvth + (size_t)i * 1024 * 1024, Wvtl + (size_t)i * 1024 * 1024,
        bv + i * 1024, vv, 1024, 1024, M, 1024);

    // scores = q k^T / sqrt(H); 50 Sinkhorn iterations -> attn
    scores_kernel<<<256, blk128, 0, stream>>>(q, kk, attn, 1.0f / 32.0f);
    sinkhorn_kernel<<<256, blk256, 0, stream>>>(attn);

    // h += bound( (0.1*ns + 0.9*attn@v + ab)*sf )
    attend_mix_kernel<<<M, blk256, 0, stream>>>(
        h, ns, attn, vv, ab + (size_t)i * 64 * 1024, sf + i * 64);

    // inter-layer LN (in place, no bf16 emission)
    if (i < 2)
      ln_kernel<<<M, blk256, 0, stream>>>(h, h, iln_g + i * 1024, iln_b + i * 1024,
                                          nullptr, nullptr);
  }

  // ---- per-agent heads (z = agent, f32 path): y = gelu(h @ W1[a] + b1[a])
  gemm_kernel<1><<<dim3(8, 2, 64), blk256, 0, stream>>>(
      h, 64 * 1024, 1024, W1, 512, 1024 * 512, b1, 512, y, 64 * 512, 512, 1024);

  // out = y @ W2[a] + b2[a]
  gemm_kernel<0><<<dim3(4, 2, 64), blk256, 0, stream>>>(
      y, 64 * 512, 512, W2, 256, 512 * 256, b2, 256, out, 64 * 256, 256, 512);
}
